// ScaledDotProductAttention_1168231104791
// MI455X (gfx1250) — matrix operand order repaired and checked by reading, not compiled, not run on hardware
//
#include <hip/hip_runtime.h>
#include <stdint.h>

// Flash-attention for: B=8, E=64, N=4096, fp32 in/out, bool mask (True -> -inf),
// softmax over keys. bf16 WMMA (v_wmma_f32_16x16x32_bf16) with fp32 accumulate.
//
// Tiling: 1 block = 4 waves = 128 threads handles one (batch, 64-query) tile.
// Each wave owns a 16-query strip. Loop over 64-key blocks with online softmax.
// Mask tile is copied global->LDS with async copies (ASYNCcnt path); K/V are
// staged through VGPRs because they convert fp32->bf16 on the way in.

#define B_SZ   8
#define E_SZ   64
#define N_SZ   4096
#define TQ     64          // queries per block
#define TK     64          // keys per iteration
#define NTHR   128

#define QLD    72          // padded LDS row strides (bf16 elements)
#define KLD    72
#define VLD    72
#define PLD    72
#define MLD    80          // mask row stride in bytes (16B-aligned for b128 async copies)

__global__ __launch_bounds__(NTHR)
void fa_bf16_wmma_kernel(const float* __restrict__ Qg,
                         const float* __restrict__ Kg,
                         const float* __restrict__ Vg,
                         const uint8_t* __restrict__ Mg,
                         float* __restrict__ Og)
{
#if defined(__HIP_DEVICE_COMPILE__)
  typedef __attribute__((ext_vector_type(16))) __bf16 v16bf;
  typedef __attribute__((ext_vector_type(8)))  float  v8f;

  __shared__ __bf16 Qs[E_SZ * QLD];   // Qs[e][q]  (bf16, pre-scaled by 1/sqrt(E))
  __shared__ __bf16 Ks[TK   * KLD];   // Ks[k][e]  (K transposed -> A-fragments row-major)
  __shared__ __bf16 Vs[E_SZ * VLD];   // Vs[e][k]
  __shared__ __bf16 Ps[TK   * PLD];   // Ps[k][q]  (staged softmax probabilities)
  __shared__ __align__(16) uint8_t Msk[TK * MLD];  // mask tile [k][q]

  const int tid  = threadIdx.x;
  const int lane = tid & 31;
  const int wave = tid >> 5;
  const int g    = (lane >> 4) & 1;   // which 16-lane half
  const int nn   = lane & 15;         // column within 16x16 C/D tile

  const int b  = blockIdx.y;
  const int q0 = blockIdx.x * TQ;             // block's first query (multiple of 64)
  const int qn = q0 + wave * 16 + nn;         // this lane's global query column
  const int qw = wave * 16 + nn;              // local query column

  const float* Qb = Qg + (size_t)b * E_SZ * N_SZ + q0;
  const float* Kb = Kg + (size_t)b * E_SZ * N_SZ;
  const float* Vb = Vg + (size_t)b * E_SZ * N_SZ;
  float*       Ob = Og + (size_t)b * E_SZ * N_SZ;

  // ---- A-fragment loader: 16x32 bf16, logical A[m][k] = base[m*ld + k].
  // ISA layout (CDNA5 ISA): lane m=lane&15; VGPR0-3 hold K=0..7 (+8 if lane>=16),
  // VGPR4-7 hold K=16..23 (+8 if lane>=16).
  auto load_a = [&](const __bf16* base, int ld) -> v16bf {
    const int m = lane & 15;
    v16bf a;
#pragma unroll
    for (int h = 0; h < 16; ++h) {
      const int k = (h & 7) + ((h >> 3) << 4) + (g << 3);
      a[h] = base[m * ld + k];
    }
    return a;
  };

  // ---- B-fragment loader: 32x16 bf16, logical Bm[k][n] = base[k*ld + n].
  // Layout: lane n=lane&15; element h holds K=(h&7)+16*(h>>3) (+8 if lane>=16), same as A.
  auto load_b = [&](const __bf16* base, int ld) -> v16bf {
    v16bf bb;
#pragma unroll
    for (int h = 0; h < 16; ++h) bb[h] = base[((h & 7) + ((h >> 3) << 4) + (g << 3)) * ld + (lane & 15)];
    return bb;
  };

  // ---- Load Q tile (E x TQ) once, fp32 -> bf16, folding in the 1/sqrt(E) scale
  // (matches reference, which scales q before the matmul).
  for (int idx = tid; idx < E_SZ * TQ; idx += NTHR) {
    const int e = idx >> 6, j = idx & 63;
    Qs[e * QLD + j] = (__bf16)(Qb[e * N_SZ + j] * 0.125f);
  }

  float m_run = -INFINITY;
  float l_run = 0.0f;
  v8f acc0 = {}, acc1 = {}, acc2 = {}, acc3 = {};

  for (int kb = 0; kb < N_SZ / TK; ++kb) {
    __syncthreads();   // previous iteration's LDS readers done before overwrite
    const int k0g = kb * TK;

    // Async-copy mask tile [64 keys][64 q-bytes] global->LDS (16B aligned both
    // sides; 2 lanes' worth per thread; no divergence so EXEC is all-ones).
    // Tracked on ASYNCcnt, drained before the publishing barrier below.
#pragma unroll
    for (int idx = tid; idx < TK * 4; idx += NTHR) {
      const int r = idx >> 2, seg = idx & 3;
      const uint64_t ga = (uint64_t)(uintptr_t)&Mg[(size_t)(k0g + r) * N_SZ + q0 + seg * 16];
      const uint32_t la = (uint32_t)(uintptr_t)&Msk[r * MLD + seg * 16];
      asm volatile("global_load_async_to_lds_b128 %0, %1, off"
                   :: "v"(la), "v"(ga) : "memory");
    }

    // Stage K (transposed) and V for this key block (coalesced global reads,
    // fp32 -> bf16 conversion through VGPRs).
    for (int idx = tid; idx < E_SZ * TK; idx += NTHR) {
      const int e = idx >> 6, i = idx & 63;
      Ks[i * KLD + e] = (__bf16)Kb[e * N_SZ + k0g + i];
      Vs[e * VLD + i] = (__bf16)Vb[e * N_SZ + k0g + i];
    }
    if (kb + 1 < N_SZ / TK) {        // global_prefetch_b8 of next block
      const int e = tid & 63, h = tid >> 6;
      __builtin_prefetch(&Kb[e * N_SZ + k0g + TK + h * 32], 0, 1);
      __builtin_prefetch(&Vb[e * N_SZ + k0g + TK + h * 32], 0, 1);
      if (h == 0) __builtin_prefetch(&Mg[(size_t)(k0g + TK + e) * N_SZ + q0], 0, 1);
    }
    asm volatile("s_wait_asynccnt 0" ::: "memory");   // mask tile landed in LDS
    __syncthreads();

    // ---- S = K^T * Qscaled : 4 key-subtiles of 16x16, E=64 in two K=32 WMMAs.
    v8f s[4];
#pragma unroll
    for (int t = 0; t < 4; ++t) {
      v8f c = {};
#pragma unroll
      for (int ee = 0; ee < 2; ++ee) {
        v16bf a  = load_a(&Ks[(t * 16) * KLD + ee * 32], KLD);
        v16bf bq = load_b(&Qs[(ee * 32) * QLD + wave * 16], QLD);
        c = __builtin_amdgcn_wmma_f32_16x16x32_bf16(false, a, false, bq,
                                                    (short)0, c, false, false);
      }
      s[t] = c;
    }

    // ---- mask + block max (C/D layout: row m = v + 8g, col = nn)
    float bm = -INFINITY;
#pragma unroll
    for (int t = 0; t < 4; ++t) {
#pragma unroll
      for (int v = 0; v < 8; ++v) {
        const int kl = t * 16 + g * 8 + v;
        float val = Msk[kl * MLD + qw] ? -INFINITY : s[t][v];
        s[t][v] = val;
        bm = fmaxf(bm, val);
      }
    }
    bm = fmaxf(bm, __shfl_xor(bm, 16, 32));          // combine lane halves (same column)
    const float m_new = fmaxf(m_run, bm);
    const float corr  = __expf(m_run - m_new);       // first iter: exp(-inf)=0

    // ---- exponentiate, accumulate column sum, stage P (bf16) to LDS
    float ls = 0.0f;
#pragma unroll
    for (int t = 0; t < 4; ++t) {
#pragma unroll
      for (int v = 0; v < 8; ++v) {
        const int kl = t * 16 + g * 8 + v;
        const float p = __expf(s[t][v] - m_new);
        ls += p;
        Ps[kl * PLD + wave * 16 + nn] = (__bf16)p;
      }
    }
    ls += __shfl_xor(ls, 16, 32);
    l_run = l_run * corr + ls;
    m_run = m_new;
    acc0 *= corr; acc1 *= corr; acc2 *= corr; acc3 *= corr;

    // ---- O += V * P : 4 E-row tiles x 2 key chunks (same-wave LDS is in-order)
#pragma unroll
    for (int kk = 0; kk < 2; ++kk) {
      v16bf bp = load_b(&Ps[(kk * 32) * PLD + wave * 16], PLD);
      v16bf a0 = load_a(&Vs[ 0 * VLD + kk * 32], VLD);
      acc0 = __builtin_amdgcn_wmma_f32_16x16x32_bf16(false, a0, false, bp, (short)0, acc0, false, false);
      v16bf a1 = load_a(&Vs[16 * VLD + kk * 32], VLD);
      acc1 = __builtin_amdgcn_wmma_f32_16x16x32_bf16(false, a1, false, bp, (short)0, acc1, false, false);
      v16bf a2 = load_a(&Vs[32 * VLD + kk * 32], VLD);
      acc2 = __builtin_amdgcn_wmma_f32_16x16x32_bf16(false, a2, false, bp, (short)0, acc2, false, false);
      v16bf a3 = load_a(&Vs[48 * VLD + kk * 32], VLD);
      acc3 = __builtin_amdgcn_wmma_f32_16x16x32_bf16(false, a3, false, bp, (short)0, acc3, false, false);
    }
  }

  // ---- epilogue: normalize and store fp32 (coalesced over q within 16 lanes)
  const float rinv = 1.0f / l_run;
#pragma unroll
  for (int v = 0; v < 8; ++v) {
    const int er = g * 8 + v;
    Ob[(size_t)( 0 + er) * N_SZ + qn] = acc0[v] * rinv;
    Ob[(size_t)(16 + er) * N_SZ + qn] = acc1[v] * rinv;
    Ob[(size_t)(32 + er) * N_SZ + qn] = acc2[v] * rinv;
    Ob[(size_t)(48 + er) * N_SZ + qn] = acc3[v] * rinv;
  }
#endif  // __HIP_DEVICE_COMPILE__
}

extern "C" void kernel_launch(void* const* d_in, const int* in_sizes, int n_in,
                              void* d_out, int out_size, void* d_ws, size_t ws_size,
                              hipStream_t stream) {
  (void)in_sizes; (void)n_in; (void)out_size; (void)d_ws; (void)ws_size;
  const float*   q = (const float*)d_in[0];
  const float*   k = (const float*)d_in[1];
  const float*   v = (const float*)d_in[2];
  const uint8_t* m = (const uint8_t*)d_in[3];  // jnp.bool_ -> 1 byte/elem
  float* out = (float*)d_out;

  dim3 grid(N_SZ / TQ, B_SZ);   // 64 query tiles x 8 batches = 512 workgroups
  fa_bf16_wmma_kernel<<<grid, NTHR, 0, stream>>>(q, k, v, m, out);
}
